// HGCNDecoder_60069412601886
// MI455X (gfx1250) — matrix operand order repaired and checked by reading, not compiled, not run on hardware
//
#include <hip/hip_runtime.h>

// ---------------------------------------------------------------------------
// HGCN layer (HypLinear -> HypAgg -> HypAct -> logmap0 decode), c = 1.0
//   x:      [N, 512] f32   (points on Poincare ball)
//   weight: [16, 512] f32
//   bias:   [16] f32
//   edge_w: [E] f32, edge_src/edge_dst: [E] int
//   out:    [N, 16] f32
// GEMM x @ W.T done with v_wmma_f32_16x16x32_f16 (one wave = one 16-row tile).
// ---------------------------------------------------------------------------

typedef __attribute__((ext_vector_type(16))) _Float16 v16h;
typedef __attribute__((ext_vector_type(8)))  float    v8f;

#define D_IN      512
#define KOUT      16
#define MIN_NORMF 1e-15f
#define BALL_MAXN (1.0f - 4e-3f)   // (1 - BALL_EPS)/sqrt(c), c=1
#define ART_CLIP  (1.0f - 1e-7f)

__device__ __forceinline__ float clipped_norm16(const float* v) {
    float s = 0.f;
#pragma unroll
    for (int j = 0; j < 16; ++j) s += v[j] * v[j];
    return fmaxf(sqrtf(s), MIN_NORMF);
}

__device__ __forceinline__ void proj16(float* v) {
    float n = clipped_norm16(v);
    if (n > BALL_MAXN) {
        float sc = BALL_MAXN / n;
#pragma unroll
        for (int j = 0; j < 16; ++j) v[j] *= sc;
    }
}

__device__ __forceinline__ float artanh_clip(float z) {
    z = fminf(z,  ART_CLIP);
    z = fmaxf(z, -ART_CLIP);
    return 0.5f * logf((1.f + z) / (1.f - z));
}

__device__ __forceinline__ void expmap0_16(float* v) {
    float n = clipped_norm16(v);
    float s = tanhf(n) / n;
#pragma unroll
    for (int j = 0; j < 16; ++j) v[j] *= s;
}

__device__ __forceinline__ void logmap0_16(float* v) {
    float n = clipped_norm16(v);
    float s = artanh_clip(n) / n;
#pragma unroll
    for (int j = 0; j < 16; ++j) v[j] *= s;
}

// ---------------------------------------------------------------------------
// Kernel 1: HypLinear via WMMA + per-row hyperbolic math -> xt = logmap0(h)
// 128 threads = 4 waves; each wave owns one 16-row tile of x.
// ---------------------------------------------------------------------------
__global__ __launch_bounds__(128)
void hgcn_hyplinear_wmma(const float* __restrict__ x,
                         const float* __restrict__ weight,
                         const float* __restrict__ bias,
                         float* __restrict__ g_xt, int N) {
    __shared__ float lds_mv[4][16][17];   // pad-17: conflict-free row reads

    const int wave = threadIdx.x >> 5;
    const int lane = threadIdx.x & 31;
    const int hf   = lane >> 4;           // half-wave select
    const int r    = lane & 15;           // row (A) / column (B) index
    const int row0 = blockIdx.x * 64 + wave * 16;

    const int rowA = row0 + r;
    const int rowL = (rowA < N) ? rowA : (N - 1);          // clamp loads
    const float* __restrict__ xrow = x + (size_t)rowL * D_IN;
    const float* __restrict__ wrow = weight + (size_t)r * D_IN + 16 * hf;

    v8f acc = {};
    float sumsq = 0.f;                     // this lane's share of ||x_row||^2

    for (int k0 = 0; k0 < D_IN; k0 += 32) {
        v16h a, b;
        // A (16x32 f16): lanes 0-15 carry K 0..15, lanes 16-31 K 16..31
#pragma unroll
        for (int i = 0; i < 8; ++i) {
            float f0 = xrow[k0 + 16 * hf + i];
            float f1 = xrow[k0 + 16 * hf + 8 + i];
            a[i]     = (_Float16)f0;
            a[8 + i] = (_Float16)f1;
            sumsq += f0 * f0 + f1 * f1;
        }
        // B (32x16 f16): B[k][n] = W[n][k]; half-waves split K 0-15 / 16-31
#pragma unroll
        for (int i = 0; i < 16; ++i)
            b[i] = (_Float16)wrow[k0 + i];

        acc = __builtin_amdgcn_wmma_f32_16x16x32_f16(
                  false, a, false, b, (short)0, acc, false, false);
    }

    // C/D layout: lanes 0-15 -> (M=v, N=lane); lanes 16-31 -> (M=v+8, N=lane-16)
#pragma unroll
    for (int v = 0; v < 8; ++v)
        lds_mv[wave][v + 8 * hf][r] = acc[v];

    // combine the two half-rows of ||x||^2 (lanes L and L^16 share a row)
    float xn2 = sumsq + __shfl_xor(sumsq, 16, 32);
    __syncthreads();

    if (hf == 0 && rowA < N) {
        float mv[16];
#pragma unroll
        for (int j = 0; j < 16; ++j) mv[j] = lds_mv[wave][r][j];

        float mxn2 = 0.f;
#pragma unroll
        for (int j = 0; j < 16; ++j) mxn2 += mv[j] * mv[j];
        float mxn = fmaxf(sqrtf(mxn2), MIN_NORMF);
        float xn  = fmaxf(sqrtf(xn2),  MIN_NORMF);

        // mobius_matvec: tanh(mxn/xn * artanh(xn)) * mx / mxn   (sqrt(c)=1)
        float t  = tanhf((mxn / xn) * artanh_clip(xn));
        float sc = t / mxn;
        float h[16];
#pragma unroll
        for (int j = 0; j < 16; ++j)
            h[j] = (mxn2 == 0.f) ? 0.f : sc * mv[j];
        proj16(h);

        // hyp_bias = proj(expmap0(bias))
        float hb[16];
#pragma unroll
        for (int j = 0; j < 16; ++j) hb[j] = bias[j];
        expmap0_16(hb);
        proj16(hb);

        // mobius_add(h, hb), c = 1
        float x2 = 0.f, y2 = 0.f, xy = 0.f;
#pragma unroll
        for (int j = 0; j < 16; ++j) {
            x2 += h[j] * h[j];  y2 += hb[j] * hb[j];  xy += h[j] * hb[j];
        }
        float cx  = 1.f + 2.f * xy + y2;
        float cy  = 1.f - x2;
        float den = fmaxf(1.f + 2.f * xy + x2 * y2, MIN_NORMF);
        float o[16];
#pragma unroll
        for (int j = 0; j < 16; ++j) o[j] = (cx * h[j] + cy * hb[j]) / den;
        proj16(o);

        // xt = logmap0(proj(h))
        logmap0_16(o);
        float* dst = g_xt + (size_t)rowA * KOUT;
#pragma unroll
        for (int j = 0; j < 16; ++j) dst[j] = o[j];
    }
}

// ---------------------------------------------------------------------------
// Kernel 2: zero the aggregation buffer (workspace is poisoned by harness)
// ---------------------------------------------------------------------------
__global__ void hgcn_zero_f32(float* __restrict__ p, int n) {
    int i = blockIdx.x * blockDim.x + threadIdx.x;
    if (i < n) p[i] = 0.f;
}

// ---------------------------------------------------------------------------
// Kernel 3: edge aggregation  agg[dst] += w * xt[src]  (one thread per edge)
// unsafeAtomicAdd -> hardware global_atomic_add_f32 (no CAS loop)
// ---------------------------------------------------------------------------
__global__ __launch_bounds__(256)
void hgcn_edge_agg(const float* __restrict__ xt,
                   const float* __restrict__ ew,
                   const int*   __restrict__ esrc,
                   const int*   __restrict__ edst,
                   float* __restrict__ agg, int E) {
    int e = blockIdx.x * blockDim.x + threadIdx.x;
    if (e >= E) return;
    int   s = esrc[e];
    int   d = edst[e];
    float w = ew[e];
    const float4* __restrict__ srow = (const float4*)(xt + (size_t)s * KOUT);
    float* __restrict__ drow = agg + (size_t)d * KOUT;
#pragma unroll
    for (int q = 0; q < 4; ++q) {
        float4 v = srow[q];
        unsafeAtomicAdd(drow + 4 * q + 0, w * v.x);
        unsafeAtomicAdd(drow + 4 * q + 1, w * v.y);
        unsafeAtomicAdd(drow + 4 * q + 2, w * v.z);
        unsafeAtomicAdd(drow + 4 * q + 3, w * v.w);
    }
}

// ---------------------------------------------------------------------------
// Kernel 4: HypAgg tail + HypAct + decode, one thread per node
//   out = logmap0(proj(expmap0(logmap0(proj(expmap0(agg))))))
// ---------------------------------------------------------------------------
__global__ __launch_bounds__(256)
void hgcn_finalize(const float* __restrict__ agg,
                   float* __restrict__ out, int N) {
    int i = blockIdx.x * blockDim.x + threadIdx.x;
    if (i >= N) return;
    float v[16];
    const float* __restrict__ a = agg + (size_t)i * KOUT;
#pragma unroll
    for (int j = 0; j < 16; ++j) v[j] = a[j];
    expmap0_16(v); proj16(v);   // h   = proj(expmap0(agg))
    logmap0_16(v);              // xt2 = logmap0(h)
    expmap0_16(v); proj16(v);   // h   = proj(expmap0(xt2))
    logmap0_16(v);              // out = logmap0(h)
    float* __restrict__ o = out + (size_t)i * KOUT;
#pragma unroll
    for (int j = 0; j < 16; ++j) o[j] = v[j];
}

// ---------------------------------------------------------------------------
extern "C" void kernel_launch(void* const* d_in, const int* in_sizes, int n_in,
                              void* d_out, int out_size, void* d_ws, size_t ws_size,
                              hipStream_t stream) {
    const float* x      = (const float*)d_in[0];   // [N, 512]
    const float* weight = (const float*)d_in[1];   // [16, 512]
    const float* bias   = (const float*)d_in[2];   // [16]
    const float* ew     = (const float*)d_in[3];   // [E]
    const int*   esrc   = (const int*)d_in[4];     // [E]  (harness: int -> const int*)
    const int*   edst   = (const int*)d_in[5];     // [E]
    float*       out    = (float*)d_out;           // [N, 16]

    const int N = in_sizes[0] / D_IN;
    const int E = in_sizes[3];

    float* g_xt  = (float*)d_ws;                     // [N,16]
    float* g_agg = g_xt + (size_t)N * KOUT;          // [N,16]

    // 1) HypLinear (WMMA GEMM + hyperbolic row math) -> xt
    hgcn_hyplinear_wmma<<<(N + 63) / 64, 128, 0, stream>>>(x, weight, bias, g_xt, N);

    // 2) zero agg, 3) sparse aggregation via fp32 hardware atomics
    const int nz = N * KOUT;
    hgcn_zero_f32<<<(nz + 255) / 256, 256, 0, stream>>>(g_agg, nz);
    hgcn_edge_agg<<<(E + 255) / 256, 256, 0, stream>>>(g_xt, ew, esrc, edst, g_agg, E);

    // 4) HypAct + decode
    hgcn_finalize<<<(N + 255) / 256, 256, 0, stream>>>(g_agg, out, N);
}
